// DividedAttention_22986664968912
// MI455X (gfx1250) — matrix operand order repaired and checked by reading, not compiled, not run on hardware
//
#include <hip/hip_runtime.h>
#include <stdint.h>

typedef __attribute__((ext_vector_type(16))) __bf16 v16bf;
typedef __attribute__((ext_vector_type(8)))  float  v8f;

#define DIM   1024
#define NH    16
#define HD    64
#define BATCH 4
#define NTOK  3137
#define NFRM  16
#define NSPA  196
#define KVLEN 197     // cls + 196 spatial
#define MROWS (BATCH*NTOK)   // 12548

static __device__ __forceinline__ unsigned short f2bf(float f){
  unsigned int u = __float_as_uint(f);
  u += 0x7fffu + ((u >> 16) & 1u);            // round-to-nearest-even
  return (unsigned short)(u >> 16);
}
static __device__ __forceinline__ float bf2f(unsigned short h){
  return __uint_as_float(((unsigned int)h) << 16);
}

union FragBF { v16bf v; uint4 q[2]; unsigned int u[8]; };

static __device__ __forceinline__ v8f wmma_bf16(v16bf a, v16bf b, v8f c){
  // D(f32 16x16) = A(bf16 16x32) * B(bf16 32x16) + C
  return __builtin_amdgcn_wmma_f32_16x16x32_bf16(false, a, false, b,
                                                 (short)0, c, false, false);
}

// ---------------------------------------------------------------- convert
__global__ void k_cvt_bf16(const float* __restrict__ s,
                           unsigned short* __restrict__ d, int n){
  int i = blockIdx.x * blockDim.x + threadIdx.x;
  if (i < n) d[i] = f2bf(s[i]);
}

// ---------------------------------------------------------------- GEMM core
// C(128x128 block) = A(M x 1024, bf16) * W^T (W is Nout x 1024, bf16)
// Software-pipelined: global->reg prefetch of tile i+1 overlaps WMMAs on
// LDS tile i; double-buffered LDS; ONE barrier per 32-wide K step.
#define LDA 40   // 128x32 tile rows padded to 40 ushorts (80B) -> conflict-free
#define TILE_USH (128*LDA)

static __device__ __forceinline__ void gemm_mainloop_128x128(
    const unsigned short* __restrict__ A,
    const unsigned short* __restrict__ W,
    unsigned short* a_lds,            // [2*TILE_USH]
    unsigned short* w_lds,            // [2*TILE_USH]
    int mBase, int nBase, v8f acc[2][4])
{
  const int tid  = threadIdx.x;
  const int lane = tid & 31, wv = tid >> 5;
  const int half = (lane >> 4) & 1, lc = lane & 15;
  const int waveM = wv >> 1, waveN = wv & 1;

  // per-thread quad slots: 4 quads of 4 bf16 for A tile, same for W tile
  int rowq[4], c4q[4];
  for (int i = 0; i < 4; i++){
    int qi = tid + i*256;
    rowq[i] = qi >> 3;
    c4q[i]  = (qi & 7) * 4;
  }

  uint2 pa[4], pw[4];
  auto fetch = [&](int kk){
    for (int i = 0; i < 4; i++){
      int gr = mBase + rowq[i];
      uint2 va; va.x = 0u; va.y = 0u;
      if (gr < MROWS) va = *(const uint2*)(A + (size_t)gr*DIM + kk + c4q[i]);
      pa[i] = va;
      int gn = nBase + rowq[i];                  // always < Nout (grid exact)
      pw[i] = *(const uint2*)(W + (size_t)gn*DIM + kk + c4q[i]);
    }
  };
  auto deposit = [&](int buf){
    unsigned short* al = a_lds + buf*TILE_USH;
    unsigned short* wl = w_lds + buf*TILE_USH;
    for (int i = 0; i < 4; i++){
      *(uint2*)(al + rowq[i]*LDA + c4q[i]) = pa[i];
      *(uint2*)(wl + rowq[i]*LDA + c4q[i]) = pw[i];
    }
  };

  fetch(0);
  deposit(0);

  for (int it = 0; it < DIM/32; it++){
    __syncthreads();                 // tile `it` visible; prev compute done
    int cur = it & 1;
    fetch(((it + 1) & (DIM/32 - 1)) * 32);       // prefetch next (wraps, dead on last)

    const unsigned short* al = a_lds + cur*TILE_USH;
    const unsigned short* wl = w_lds + cur*TILE_USH;
    FragBF afr[2], bfr[4];
    for (int mt = 0; mt < 2; mt++){
      // A 16x32: lane row = lc; K = [16*half, 16*half+16) contiguous
      const unsigned short* p = al + (waveM*32 + mt*16 + lc)*LDA + half*16;
      afr[mt].q[0] = *(const uint4*)(p);
      afr[mt].q[1] = *(const uint4*)(p + 8);
    }
    for (int nt = 0; nt < 4; nt++){
      // B 32x16: lane col = lc; K = [16*half, 16*half+16) contiguous
      const unsigned short* p = wl + (waveN*64 + nt*16 + lc)*LDA + half*16;
      bfr[nt].q[0] = *(const uint4*)(p);
      bfr[nt].q[1] = *(const uint4*)(p + 8);
    }
    for (int mt = 0; mt < 2; mt++)
      for (int nt = 0; nt < 4; nt++)
        acc[mt][nt] = wmma_bf16(afr[mt].v, bfr[nt].v, acc[mt][nt]);

    deposit(cur ^ 1);                // safe: everyone left buf cur^1 at barrier
  }
}

// ---------------------------------------------------------------- QKV GEMM
__global__ __launch_bounds__(256) void gemm_qkv(
    const unsigned short* __restrict__ A,
    const unsigned short* __restrict__ W,
    const float* __restrict__ bias,
    unsigned short* __restrict__ qb,
    unsigned short* __restrict__ kb,
    unsigned short* __restrict__ vb)
{
  __shared__ unsigned short a_lds[2*TILE_USH];
  __shared__ unsigned short w_lds[2*TILE_USH];
  const int tid  = threadIdx.x;
  const int lane = tid & 31, wv = tid >> 5;
  const int half = (lane >> 4) & 1, lc = lane & 15;
  const int mBase = blockIdx.x * 128, nBase = blockIdx.y * 128;
  const int waveM = wv >> 1, waveN = wv & 1;

  v8f zero = {0.f,0.f,0.f,0.f,0.f,0.f,0.f,0.f};
  v8f acc[2][4];
  for (int i = 0; i < 2; i++) for (int j = 0; j < 4; j++) acc[i][j] = zero;

  gemm_mainloop_128x128(A, W, a_lds, w_lds, mBase, nBase, acc);

  for (int mt = 0; mt < 2; mt++)
    for (int nt = 0; nt < 4; nt++){
      int col = nBase + waveN*64 + nt*16 + lc;
      float bcol = bias[col];
      int t  = col >> 10, cc = col & 1023;
      int hh = cc >> 6,  dd = cc & 63;
      unsigned short* dst = (t == 0) ? qb : ((t == 1) ? kb : vb);
      float scl = (t == 0) ? 0.125f : 1.0f;      // q *= hd^-0.5
      for (int v = 0; v < 8; v++){
        int gr = mBase + waveM*32 + mt*16 + v + 8*half;
        if (gr >= MROWS) continue;
        int bb = gr / NTOK, nn = gr - bb*NTOK;
        float val = (acc[mt][nt][v] + bcol) * scl;
        dst[((size_t)(bb*NH + hh)*NTOK + nn)*HD + dd] = f2bf(val);
      }
    }
}

// ---------------------------------------------------------------- proj GEMM
__global__ __launch_bounds__(256) void gemm_proj(
    const unsigned short* __restrict__ A,   // (M,1024) bf16 attn-out
    const unsigned short* __restrict__ W,   // (1024,1024) bf16
    const float* __restrict__ bias,
    float* __restrict__ out)
{
  __shared__ unsigned short a_lds[2*TILE_USH];
  __shared__ unsigned short w_lds[2*TILE_USH];
  const int tid  = threadIdx.x;
  const int lane = tid & 31, wv = tid >> 5;
  const int half = (lane >> 4) & 1, lc = lane & 15;
  const int mBase = blockIdx.x * 128, nBase = blockIdx.y * 128;
  const int waveM = wv >> 1, waveN = wv & 1;

  v8f zero = {0.f,0.f,0.f,0.f,0.f,0.f,0.f,0.f};
  v8f acc[2][4];
  for (int i = 0; i < 2; i++) for (int j = 0; j < 4; j++) acc[i][j] = zero;

  gemm_mainloop_128x128(A, W, a_lds, w_lds, mBase, nBase, acc);

  for (int mt = 0; mt < 2; mt++)
    for (int nt = 0; nt < 4; nt++){
      int col = nBase + waveN*64 + nt*16 + lc;
      float bcol = bias[col];
      for (int v = 0; v < 8; v++){
        int gr = mBase + waveM*32 + mt*16 + v + 8*half;
        if (gr >= MROWS) continue;
        out[(size_t)gr*DIM + col] = acc[mt][nt][v] + bcol;
      }
    }
}

// ---------------------------------------------------------------- spatial attn
// One block per (b,h,f). 13 waves; wave w owns q rows [16w,16w+16).
// kv = [cls, 196 frame tokens], padded to 224 = 7 chunks of 32.
#define KLDS 72   // k chunk row stride (ushorts), 16B aligned + bank-rotated
#define VLDS 40   // transposed V stride
#define PLDS 40   // per-wave P tile stride

__global__ __launch_bounds__(416) void attn_spatial(
    const unsigned short* __restrict__ qb,
    const unsigned short* __restrict__ kb,
    const unsigned short* __restrict__ vb,
    unsigned short* __restrict__ ao)
{
  __shared__ unsigned short k_lds[32*KLDS];
  __shared__ unsigned short vt_lds[64*VLDS];
  __shared__ unsigned short p_lds[13*16*PLDS];

  const int tid  = threadIdx.x;
  const int lane = tid & 31, w = tid >> 5;
  const int half = (lane >> 4) & 1, lc = lane & 15;
  const int bid = blockIdx.x;
  const int f = bid % NFRM;
  const int h = (bid / NFRM) % NH;
  const int b = bid / (NFRM*NH);
  const size_t baseQ = (size_t)(b*NH + h) * NTOK * HD;

  // q A-fragments for the two K-steps (d 0..31, 32..63), kept in registers
  int jq = 16*w + lc; if (jq > NSPA-1) jq = NSPA-1;
  const unsigned short* qrow = qb + baseQ + (size_t)(1 + f*NSPA + jq)*HD;
  FragBF qf[2];
  for (int s = 0; s < 2; s++){
    const unsigned short* p = qrow + s*32 + half*16;
    qf[s].q[0] = *(const uint4*)(p);
    qf[s].q[1] = *(const uint4*)(p + 8);
  }

  v8f zero = {0.f,0.f,0.f,0.f,0.f,0.f,0.f,0.f};
  float mrow[8], lrow[8];
  v8f o[4];
  for (int v = 0; v < 8; v++){ mrow[v] = -1e30f; lrow[v] = 0.f; }
  for (int nt = 0; nt < 4; nt++) o[nt] = zero;

  unsigned short* pst = p_lds + w*16*PLDS;

  for (int ch = 0; ch < 7; ch++){
    __syncthreads();
    for (int idx = tid; idx < 32*64; idx += 416){
      int row = idx >> 6, d = idx & 63;
      int kv = ch*32 + row;
      unsigned short kvv = 0, vvv = 0;
      if (kv <= NSPA){
        int tok = (kv == 0) ? 0 : (f*NSPA + kv);
        kvv = kb[baseQ + (size_t)tok*HD + d];
        vvv = vb[baseQ + (size_t)tok*HD + d];
      }
      k_lds[row*KLDS + d]  = kvv;     // row-major for S B-frags
      vt_lds[d*VLDS + row] = vvv;     // transposed for PV B-frags
    }
    __syncthreads();

    // S = q * k^T : two 16-col kv tiles, K=64 in two WMMA steps
    v8f s0 = zero, s1 = zero;
    for (int s = 0; s < 2; s++){
      FragBF bk0, bk1;
      const unsigned short* p0 = k_lds + lc*KLDS        + s*32 + half*16;
      const unsigned short* p1 = k_lds + (16 + lc)*KLDS + s*32 + half*16;
      bk0.q[0] = *(const uint4*)(p0); bk0.q[1] = *(const uint4*)(p0 + 8);
      bk1.q[0] = *(const uint4*)(p1); bk1.q[1] = *(const uint4*)(p1 + 8);
      s0 = wmma_bf16(qf[s].v, bk0.v, s0);
      s1 = wmma_bf16(qf[s].v, bk1.v, s1);
    }

    // online softmax; rows live per-VGPR, columns across the 16-lane half
    int kv0 = ch*32 + lc, kv1 = kv0 + 16;
    bool ok0 = kv0 < KVLEN, ok1 = kv1 < KVLEN;
    float sc[8];
    for (int v = 0; v < 8; v++){
      float a  = ok0 ? s0[v] : -1e30f;
      float bbv= ok1 ? s1[v] : -1e30f;
      float mv = fmaxf(a, bbv);
      for (int off = 1; off < 16; off <<= 1)
        mv = fmaxf(mv, __shfl_xor(mv, off, 16));
      float mnew  = fmaxf(mrow[v], mv);
      float scale = __expf(mrow[v] - mnew);
      float p0 = ok0 ? __expf(s0[v] - mnew) : 0.f;
      float p1 = ok1 ? __expf(s1[v] - mnew) : 0.f;
      float rs = p0 + p1;
      for (int off = 1; off < 16; off <<= 1)
        rs += __shfl_xor(rs, off, 16);
      lrow[v] = lrow[v]*scale + rs;
      mrow[v] = mnew;
      sc[v] = scale;
      int prow = v + 8*half;                     // C/D layout row
      pst[prow*PLDS + lc]      = f2bf(p0);
      pst[prow*PLDS + 16 + lc] = f2bf(p1);
    }
    for (int nt = 0; nt < 4; nt++)
      for (int v = 0; v < 8; v++)
        o[nt][v] *= sc[v];

    asm volatile("s_wait_dscnt 0" ::: "memory"); // P write -> A-frag read (same wave)
    FragBF pf;
    const unsigned short* pp = pst + lc*PLDS + half*16;
    pf.q[0] = *(const uint4*)(pp);
    pf.q[1] = *(const uint4*)(pp + 8);

    for (int nt = 0; nt < 4; nt++){
      FragBF bvf;
      const unsigned short* vp = vt_lds + (nt*16 + lc)*VLDS + half*16;
      bvf.q[0] = *(const uint4*)(vp);
      bvf.q[1] = *(const uint4*)(vp + 8);
      o[nt] = wmma_bf16(pf.v, bvf.v, o[nt]);
    }
  }

  for (int v = 0; v < 8; v++){
    int j = 16*w + v + 8*half;
    if (j >= NSPA) continue;
    float inv = 1.0f / lrow[v];
    size_t tok = 1 + (size_t)f*NSPA + j;
    unsigned short* dst = ao + ((size_t)b*NTOK + tok)*DIM + h*HD;
    for (int nt = 0; nt < 4; nt++)
      dst[nt*16 + lc] = f2bf(o[nt][v] * inv);
  }
}

// ---------------------------------------------------------------- cls attn
__global__ __launch_bounds__(256) void attn_cls(
    const unsigned short* __restrict__ qb,
    const unsigned short* __restrict__ kb,
    const unsigned short* __restrict__ vb,
    unsigned short* __restrict__ ao)
{
  __shared__ float logits[NTOK];
  __shared__ float red[256];
  __shared__ float q_sh[HD];
  const int tid = threadIdx.x;
  const int bh = blockIdx.x;
  const int b = bh / NH, h = bh % NH;
  const size_t base = (size_t)bh * NTOK * HD;
  if (tid < HD) q_sh[tid] = bf2f(qb[base + tid]);   // cls q, already scaled
  __syncthreads();
  float lmax = -1e30f;
  for (int t = tid; t < NTOK; t += 256){
    const unsigned short* kr = kb + base + (size_t)t*HD;
    float s = 0.f;
    for (int d = 0; d < HD; d++) s += q_sh[d] * bf2f(kr[d]);
    logits[t] = s;
    lmax = fmaxf(lmax, s);
  }
  red[tid] = lmax; __syncthreads();
  for (int st = 128; st > 0; st >>= 1){
    if (tid < st) red[tid] = fmaxf(red[tid], red[tid + st]);
    __syncthreads();
  }
  float M = red[0]; __syncthreads();
  float lsum = 0.f;
  for (int t = tid; t < NTOK; t += 256){
    float p = __expf(logits[t] - M);
    logits[t] = p; lsum += p;
  }
  red[tid] = lsum; __syncthreads();
  for (int st = 128; st > 0; st >>= 1){
    if (tid < st) red[tid] += red[tid + st];
    __syncthreads();
  }
  float L = red[0]; __syncthreads();
  if (tid < HD){
    float acc = 0.f;
    for (int t = 0; t < NTOK; t++)
      acc += logits[t] * bf2f(vb[base + (size_t)t*HD + tid]);
    ao[(size_t)b*NTOK*DIM + h*HD + tid] = f2bf(acc / L);
  }
}

// ---------------------------------------------------------------- launch
extern "C" void kernel_launch(void* const* d_in, const int* in_sizes, int n_in,
                              void* d_out, int out_size, void* d_ws, size_t ws_size,
                              hipStream_t stream)
{
  (void)in_sizes; (void)n_in; (void)out_size; (void)ws_size;
  const float* x      = (const float*)d_in[0];
  const float* qkv_w  = (const float*)d_in[1];
  const float* qkv_b  = (const float*)d_in[2];
  const float* proj_w = (const float*)d_in[3];
  const float* proj_b = (const float*)d_in[4];
  float* out = (float*)d_out;

  char* ws = (char*)d_ws;
  size_t off = 0;
  auto alloc = [&](size_t bytes)->void*{
    void* p = ws + off; off += (bytes + 255) & ~(size_t)255; return p;
  };
  unsigned short* xb   = (unsigned short*)alloc((size_t)MROWS*DIM*2);
  unsigned short* wq   = (unsigned short*)alloc((size_t)3*DIM*DIM*2);
  unsigned short* wp   = (unsigned short*)alloc((size_t)DIM*DIM*2);
  const size_t hv = (size_t)BATCH*NH*NTOK*HD;
  unsigned short* qbuf = (unsigned short*)alloc(hv*2);
  unsigned short* kbuf = (unsigned short*)alloc(hv*2);
  unsigned short* vbuf = (unsigned short*)alloc(hv*2);
  unsigned short* ao   = (unsigned short*)alloc((size_t)MROWS*DIM*2);

  int nx = MROWS*DIM, nwq = 3*DIM*DIM, nwp = DIM*DIM;
  k_cvt_bf16<<<(nx  + 255)/256, 256, 0, stream>>>(x,      xb, nx);
  k_cvt_bf16<<<(nwq + 255)/256, 256, 0, stream>>>(qkv_w,  wq, nwq);
  k_cvt_bf16<<<(nwp + 255)/256, 256, 0, stream>>>(proj_w, wp, nwp);

  gemm_qkv<<<dim3((MROWS + 127)/128, 3*DIM/128), 256, 0, stream>>>(
      xb, wq, qkv_b, qbuf, kbuf, vbuf);

  attn_spatial<<<BATCH*NH*NFRM, 416, 0, stream>>>(qbuf, kbuf, vbuf, ao);
  attn_cls<<<BATCH*NH, 256, 0, stream>>>(qbuf, kbuf, vbuf, ao);

  gemm_proj<<<dim3((MROWS + 127)/128, DIM/128), 256, 0, stream>>>(
      ao, wp, proj_b, out);
}
